// _GRU_57406532878985
// MI455X (gfx1250) — matrix operand order repaired and checked by reading, not compiled, not run on hardware
//
#include <hip/hip_runtime.h>

// ---------------------------------------------------------------------------
// GRU-ODE scan on MI455X (gfx1250): WMMA f16 batched GRU.
// B=1024, T=512, HIDDEN=160, GRU_IN=32, gates=480.
// 64 workgroups x 320 threads (10 wave32 waves). Each WG owns 16 batch rows;
// wave i owns hidden tile i (16 hidden units) and its 3 gate column tiles.
// Weights pre-packed to WMMA B-operand layout (f16), resident in VGPRs.
// f32 hidden state lives in registers (owner-lane invariant); only the f16
// mirror used as the WMMA A operand goes through LDS. Per-step obs/snapshot
// flags are packed into one 32-bit LDS word. 3 barriers per step.
// ---------------------------------------------------------------------------

typedef __attribute__((ext_vector_type(16))) _Float16 v16h;
typedef __attribute__((ext_vector_type(8)))  _Float16 v8h;
typedef __attribute__((ext_vector_type(8)))  float    v8f;

#define HIDDEN   160
#define GRU_IN   32
#define OUT_N    32
#define BATCH    1024
#define TSTEPS   512
#define CCH      65
#define MROWS    16
#define WAVES    10
#define BLOCK    (WAVES * 32)
#define NKH      5            // K-tiles over HIDDEN (160/32)

// packed-weight layout in d_ws (halfs):
//   ih tile (g,i):   (g*10+i)*512
//   hh tile (g,i,k): 30*512 + ((g*10+i)*5 + k)*512
// within a 512-half tile: lane = n + 16*(q>>4), halfIdx = q&15  (B 32x16 f16)

__global__ void pack_weights(const float* __restrict__ w_ih,
                             const float* __restrict__ w_hh,
                             _Float16* __restrict__ ws) {
    int id = blockIdx.x * blockDim.x + threadIdx.x;
    const int total = 180 * 512;
    if (id >= total) return;
    int tile   = id >> 9;
    int within = id & 511;
    int lane   = within >> 4;
    int h      = within & 15;
    int n      = lane & 15;
    int q      = h + ((lane >> 4) << 4);     // K index within the 32-row K tile
    float v;
    if (tile < 30) {                          // w_ih (480 x 32)
        int g = tile / 10, i = tile % 10;
        int c = g * HIDDEN + i * 16 + n;
        v = w_ih[c * GRU_IN + q];
    } else {                                  // w_hh (480 x 160)
        int t2 = tile - 30;
        int g = t2 / 50;
        int i = (t2 / 5) % 10;
        int k = t2 % 5;
        int c = g * HIDDEN + i * 16 + n;
        v = w_hh[c * HIDDEN + k * 32 + q];
    }
    ws[id] = (_Float16)v;
}

__device__ __forceinline__ float fast_rcp(float x) {
    return __builtin_amdgcn_rcpf(x);
}
__device__ __forceinline__ float sigmoid_f(float x) {
    return fast_rcp(1.0f + __expf(-x));       // v_exp_f32 + v_rcp_f32
}
__device__ __forceinline__ float tanh_f(float x) {
    return 1.0f - 2.0f * fast_rcp(__expf(2.0f * x) + 1.0f);
}
__device__ __forceinline__ v8f wmma_f16(v16h a, v16h b, v8f c) {
    return __builtin_amdgcn_wmma_f32_16x16x32_f16(false, a, false, b,
                                                  (short)0, c, false, false);
}
__device__ __forceinline__ unsigned wave_ballot32(bool p) {
#if __has_builtin(__builtin_amdgcn_ballot_w32)
    return __builtin_amdgcn_ballot_w32(p);
#else
    return (unsigned)__ballot(p);
#endif
}

__global__ __launch_bounds__(BLOCK, 1)
void gru_scan(const float* __restrict__ times,
              const float* __restrict__ X,
              const int*   __restrict__ final_index,
              const float* __restrict__ b_ih,
              const float* __restrict__ b_hh,
              const _Float16* __restrict__ wpk,
              const float* __restrict__ lin_w,
              const float* __restrict__ lin_b,
              float* __restrict__ out) {
    __shared__ __align__(16) float    XT[2][MROWS][68];   // double-buffered X tile
    __shared__ __align__(16) _Float16 h16[MROWS][176];    // f16 hidden (A operand)
    __shared__ __align__(16) _Float16 x16s[MROWS][32];    // f16 GRU input
    __shared__ __align__(16) float    hfin[MROWS][HIDDEN];
    __shared__ __align__(16) float    timesS[TSTEPS];
    __shared__ float dts[MROWS];
    __shared__ int   fidxS[MROWS];
    __shared__ int   masksS[2];   // bits 0..15: obs per row; 16..31: snapshot

    const int tid    = threadIdx.x;
    const int wave   = tid >> 5;
    const int lane   = tid & 31;
    const int wgBase = blockIdx.x * MROWS;

    // ---- init ----
    for (int e = tid; e < MROWS * 176; e += BLOCK) ((_Float16*)h16)[e] = (_Float16)0.0f;
    for (int e = tid; e < TSTEPS;      e += BLOCK) timesS[e] = times[e];
    if (tid < MROWS) {
        dts[tid]   = 0.0f;
        fidxS[tid] = final_index[wgBase + tid];
    }
    if (tid == 0) { masksS[0] = 0; masksS[1] = 0; }
    for (int e = tid; e < MROWS * CCH; e += BLOCK) {
        int r = e / CCH, c = e % CCH;
        XT[0][r][c] = X[((size_t)(wgBase + r) * TSTEPS) * CCH + c];
        XT[1][r][c] = 0.0f;   // "previous" of t=0 (no delta subtraction)
    }

    // ---- load this wave's 18 B-operand tiles into registers (stay resident) ----
    const int i = wave;          // hidden tile id
    v16h Bih[3];
    v16h Bhh[3][NKH];
#pragma unroll
    for (int g = 0; g < 3; ++g)
        Bih[g] = *(const v16h*)(wpk + ((g * 10 + i) * 512 + lane * 16));
#pragma unroll
    for (int g = 0; g < 3; ++g)
#pragma unroll
        for (int k = 0; k < NKH; ++k)
            Bhh[g][k] = *(const v16h*)(wpk + (30 * 512 + ((g * 10 + i) * NKH + k) * 512 + lane * 16));

    const int n    = lane & 15;
    const int colh = i * 16 + n;          // this lane's hidden-unit column
    const int hi8  = (lane >> 4) << 3;    // 0 / 8: A half-row select AND C M-offset
    const int arow = lane & 15;           // A-operand batch row

    // loop-invariant per-lane biases (folded into epilogue)
    const float br  = b_ih[colh]       + b_hh[colh];
    const float bz  = b_ih[160 + colh] + b_hh[160 + colh];
    const float bni = b_ih[320 + colh];
    const float bnh = b_hh[320 + colh];

    // register-resident f32 hidden state + final-index snapshot (owner lane
    // of (M = r8+hi8, col = colh) never changes)
    float hReg[8], hfinReg[8];
#pragma unroll
    for (int r8 = 0; r8 < 8; ++r8) { hReg[r8] = 0.0f; hfinReg[r8] = 0.0f; }

    __syncthreads();

    // ---- sequential scan ----
    for (int t = 0; t < TSTEPS; ++t) {
        const int cur = t & 1, nxt = cur ^ 1;

        // phase 1 (waves 0..7, 2 rows/wave): deltas, obs via in-wave ballot,
        // input conversion, dt update, packed obs|snapshot mask
        if (wave < 8) {
            int row = 2 * wave + (lane >> 4);
            int q   = lane & 15;
            float d1 = XT[cur][row][1 + q]  - XT[nxt][row][1 + q];
            float d2 = XT[cur][row][17 + q] - XT[nxt][row][17 + q];
            unsigned bal = wave_ballot32(d1 > 0.5f || d2 > 0.5f);
            x16s[row][q]      = (_Float16)XT[cur][row][33 + q];
            x16s[row][q + 16] = (_Float16)XT[cur][row][49 + q];
            if (q == 0) {   // lanes 0 and 16 finalize their row
                bool o = ((lane < 16) ? (bal & 0xFFFFu) : (bal >> 16)) != 0u;
                float dtv = dts[row];
                x16s[row][0] = (_Float16)(XT[cur][row][33] + dtv);   // x[0] += dt
                float tsub = timesS[t == 0 ? 0 : t - 1];
                float d0   = XT[cur][row][0] - tsub;
                if (!o) dts[row] = dtv + d0;   // per-element form of any_obs gating
                unsigned mb = (o ? (1u << row) : 0u) |
                              ((fidxS[row] == t) ? (1u << (16 + row)) : 0u);
                if (mb) atomicOr(&masksS[cur], (int)mb);
            }
        }
        if (tid == BLOCK - 1) masksS[nxt] = 0;   // reset other buffer for t+1
        __syncthreads();

        // phase 2: prefetch next X tile (overlaps WMMA work)
        float pf[4];
        const bool havePf = (t + 1 < TSTEPS);
        if (havePf) {
#pragma unroll
            for (int j = 0; j < 4; ++j) {
                int e = tid + j * BLOCK;
                if (e < MROWS * CCH) {
                    int r = e / CCH, c = e % CCH;
                    pf[j] = X[((size_t)(wgBase + r) * TSTEPS + (t + 1)) * CCH + c];
                }
            }
        }

        // accumulators: WMMA inline-0 C operand
        v8f accR = {}, accZ = {}, accNI = {}, accNH = {};

        // input matmul: A = x16 (16x32 f16), one K tile
        {
            union { v16h v; v8h h[2]; } u;
            u.h[0] = *(const v8h*)&x16s[arow][2 * hi8];
            u.h[1] = *(const v8h*)&x16s[arow][2 * hi8 + 8];
            accR  = wmma_f16(u.v, Bih[0], accR);
            accZ  = wmma_f16(u.v, Bih[1], accZ);
            accNI = wmma_f16(u.v, Bih[2], accNI);
        }
        // hidden matmul: A = h16, 5 K tiles
#pragma unroll
        for (int k = 0; k < NKH; ++k) {
            union { v16h v; v8h h[2]; } u;
            u.h[0] = *(const v8h*)&h16[arow][k * 32 + 2 * hi8];
            u.h[1] = *(const v8h*)&h16[arow][k * 32 + 2 * hi8 + 8];
            accR  = wmma_f16(u.v, Bhh[0][k], accR);
            accZ  = wmma_f16(u.v, Bhh[1][k], accZ);
            accNH = wmma_f16(u.v, Bhh[2][k], accNH);
        }

        __syncthreads();   // all waves done reading h16/x16 before state update

        // epilogue: gates -> new h (register-resident), one packed-mask load
        const unsigned m = (unsigned)masksS[cur];
#pragma unroll
        for (int r8 = 0; r8 < 8; ++r8) {
            const int M = r8 + hi8;              // C layout: lanes>=16 hold M=8..15
            float gr = sigmoid_f(accR[r8] + br);
            float gz = sigmoid_f(accZ[r8] + bz);
            float gn = tanh_f((accNI[r8] + bni) + gr * (accNH[r8] + bnh));
            float hOld = hReg[r8];
            float hNew = gz * (hOld - gn) + gn;  // (1-z)*n + z*h
            float hUpd = ((m >> M) & 1u) ? hNew : hOld;
            hReg[r8] = hUpd;
            h16[M][colh] = (_Float16)hUpd;
            if ((m >> (16 + M)) & 1u) hfinReg[r8] = hUpd;
        }

        // commit prefetched X[t+1] into the (now free) other buffer
        if (havePf) {
#pragma unroll
            for (int j = 0; j < 4; ++j) {
                int e = tid + j * BLOCK;
                if (e < MROWS * CCH) {
                    int r = e / CCH, c = e % CCH;
                    XT[nxt][r][c] = pf[j];
                }
            }
        }
        __syncthreads();
    }

    // ---- spill snapshots to LDS once, then tiny f32 linear ----
#pragma unroll
    for (int r8 = 0; r8 < 8; ++r8) hfin[r8 + hi8][colh] = hfinReg[r8];
    __syncthreads();

    for (int p = tid; p < MROWS * OUT_N; p += BLOCK) {
        int r = p >> 5, o = p & 31;
        float acc = lin_b[o];
        for (int kk = 0; kk < HIDDEN; ++kk)
            acc += hfin[r][kk] * lin_w[o * HIDDEN + kk];
        out[(size_t)(wgBase + r) * OUT_N + o] = acc;
    }
}

extern "C" void kernel_launch(void* const* d_in, const int* in_sizes, int n_in,
                              void* d_out, int out_size, void* d_ws, size_t ws_size,
                              hipStream_t stream) {
    const float* times = (const float*)d_in[0];
    const float* X     = (const float*)d_in[1];
    const int*   fidx  = (const int*)  d_in[2];
    const float* w_ih  = (const float*)d_in[3];
    const float* w_hh  = (const float*)d_in[4];
    const float* b_ih  = (const float*)d_in[5];
    const float* b_hh  = (const float*)d_in[6];
    const float* lin_w = (const float*)d_in[7];
    const float* lin_b = (const float*)d_in[8];
    _Float16* wpk = (_Float16*)d_ws;          // 180 tiles * 512 halfs = 184,320 B

    const int total = 180 * 512;
    pack_weights<<<(total + 255) / 256, 256, 0, stream>>>(w_ih, w_hh, wpk);
    gru_scan<<<BATCH / MROWS, BLOCK, 0, stream>>>(times, X, fidx, b_ih, b_hh,
                                                  wpk, lin_w, lin_b, (float*)d_out);
}
